// RNN_Decoder_56590489092320
// MI455X (gfx1250) — matrix operand order repaired and checked by reading, not compiled, not run on hardware
//
#include <hip/hip_runtime.h>
#include <cstdint>

#define BB    64
#define SS    128
#define DENC  1024
#define HH    512
#define EE    512
#define VV    32000
#define TSTEPS 31   // max_length_trg - 1

typedef __attribute__((ext_vector_type(16))) __bf16 v16bf;
typedef __attribute__((ext_vector_type(8)))  float  v8f;

union bfrag { v16bf v; uint4 u[2]; };

// fp32 -> bf16 round-to-nearest-even (integer path)
__device__ __forceinline__ unsigned short f2bf_u(float f) {
    union { float f; unsigned u; } v; v.f = f;
    return (unsigned short)((v.u + 0x7FFFu + ((v.u >> 16) & 1u)) >> 16);
}

// packed fp32 -> bf16 conversion, 4 elements/thread (n must be a multiple of 4)
__global__ void __launch_bounds__(256)
f32_to_bf16(const float* __restrict__ in, unsigned short* __restrict__ outp, int n) {
    int i4 = (blockIdx.x * blockDim.x + threadIdx.x) * 4;
    if (i4 >= n) return;
    float4 v = *(const float4*)(in + i4);
    uint2 r;
    r.x = (unsigned)f2bf_u(v.x) | ((unsigned)f2bf_u(v.y) << 16);
    r.y = (unsigned)f2bf_u(v.z) | ((unsigned)f2bf_u(v.w) << 16);
    *(uint2*)(outp + i4) = r;
}

// A fragment: 16x32 bf16 (ISA 7.12.2). lane&15 = M row; lanes 0-15 hold
// K = k..k+7 (v0-3) and k+16..k+23 (v4-7); lanes 16-31 hold k+8.. / k+24..
__device__ __forceinline__ v16bf load_a_bf(const __bf16* __restrict__ A, int lda,
                                           int m0, int k, int lane) {
    const __bf16* p = A + (size_t)(m0 + (lane & 15)) * lda + k + ((lane >> 4) << 3);
    bfrag f;
    f.u[0] = *(const uint4*)p;
    f.u[1] = *(const uint4*)(p + 16);
    return f.v;
}

// B fragment: 32x16 bf16 = W[N,K]^T. lane&15 = N col; lanes 0-15 carry
// K=k..k+7 / k+16..k+23, lanes 16-31 carry K=k+8..k+15 / k+24..k+31.
__device__ __forceinline__ v16bf load_b_bf(const __bf16* __restrict__ W, int ldw,
                                           int n0, int k, int lane) {
    const __bf16* p = W + (size_t)(n0 + (lane & 15)) * ldw + k + ((lane >> 4) << 3);
    bfrag f;
    f.u[0] = *(const uint4*)p;
    f.u[1] = *(const uint4*)(p + 16);
    return f.v;
}

// Software-pipelined K-loop: prefetch iteration k+1's fragments before the
// WMMAs of iteration k, so loads return in-order behind a partial
// s_wait_loadcnt instead of a full drain.
__device__ __forceinline__ void gemm_acc_pipelined(
    const __bf16* __restrict__ A, const __bf16* __restrict__ W, int K,
    int m0, int n0, int lane, v8f acc[4])
{
    v16bf a  = load_a_bf(A, K, m0, 0, lane);
    v16bf b0 = load_b_bf(W, K, n0,      0, lane);
    v16bf b1 = load_b_bf(W, K, n0 + 16, 0, lane);
    v16bf b2 = load_b_bf(W, K, n0 + 32, 0, lane);
    v16bf b3 = load_b_bf(W, K, n0 + 48, 0, lane);
    for (int k = 32; k < K; k += 32) {
        v16bf an = load_a_bf(A, K, m0, k, lane);
        v16bf c0 = load_b_bf(W, K, n0,      k, lane);
        v16bf c1 = load_b_bf(W, K, n0 + 16, k, lane);
        v16bf c2 = load_b_bf(W, K, n0 + 32, k, lane);
        v16bf c3 = load_b_bf(W, K, n0 + 48, k, lane);
        acc[0] = __builtin_amdgcn_wmma_f32_16x16x32_bf16(false, a, false, b0, (short)0, acc[0], false, false);
        acc[1] = __builtin_amdgcn_wmma_f32_16x16x32_bf16(false, a, false, b1, (short)0, acc[1], false, false);
        acc[2] = __builtin_amdgcn_wmma_f32_16x16x32_bf16(false, a, false, b2, (short)0, acc[2], false, false);
        acc[3] = __builtin_amdgcn_wmma_f32_16x16x32_bf16(false, a, false, b3, (short)0, acc[3], false, false);
        a = an; b0 = c0; b1 = c1; b2 = c2; b3 = c3;
    }
    acc[0] = __builtin_amdgcn_wmma_f32_16x16x32_bf16(false, a, false, b0, (short)0, acc[0], false, false);
    acc[1] = __builtin_amdgcn_wmma_f32_16x16x32_bf16(false, a, false, b1, (short)0, acc[1], false, false);
    acc[2] = __builtin_amdgcn_wmma_f32_16x16x32_bf16(false, a, false, b2, (short)0, acc[2], false, false);
    acc[3] = __builtin_amdgcn_wmma_f32_16x16x32_bf16(false, a, false, b3, (short)0, acc[3], false, false);
}

// C = A0[M,K0] @ W0[N,K0]^T (+ A1[M,K1] @ W1[N,K1]^T) + bias   (bf16 in, f32 out)
// block = 128 threads (4 waves) -> 64x64 C tile; wave w -> rows [m0, m0+16).
__global__ void __launch_bounds__(128)
gemm_bf16_wmma(const __bf16* __restrict__ A0, const __bf16* __restrict__ W0, int K0,
               const __bf16* __restrict__ A1, const __bf16* __restrict__ W1, int K1,
               const float* __restrict__ bias,
               float* __restrict__ C, long cstride_m)
{
    const int lane = threadIdx.x & 31;
    const int wave = threadIdx.x >> 5;
    const int m0 = blockIdx.y * 64 + wave * 16;
    const int n0 = blockIdx.x * 64;

    v8f acc[4] = {};

    gemm_acc_pipelined(A0, W0, K0, m0, n0, lane, acc);
    if (A1) gemm_acc_pipelined(A1, W1, K1, m0, n0, lane, acc);

    // C/D layout: lane -> col n0+16s+(lane&15); vgpr r -> row m0 + r + 8*(lane>>4)
#pragma unroll
    for (int s = 0; s < 4; ++s) {
        int col = n0 + 16 * s + (lane & 15);
        float bv = bias ? bias[col] : 0.0f;
#pragma unroll
        for (int r = 0; r < 8; ++r) {
            int row = m0 + r + ((lane >> 4) << 3);
            C[(size_t)row * cstride_m + col] = acc[s][r] + bv;
        }
    }
}

__global__ void init_state(float* h0, float* c0, float* h1, float* c1,
                           unsigned short* h0b, unsigned short* h1b, int* pred) {
    int i = blockIdx.x * blockDim.x + threadIdx.x;
    if (i < BB * HH) {
        h0[i] = 0.f; c0[i] = 0.f; h1[i] = 0.f; c1[i] = 0.f;
        h0b[i] = 0; h1b[i] = 0;
    }
    if (i < BB) pred[i] = 1;  // START token
}

// one block per batch row: scores -> masked softmax -> context; embeds pred.
// Writes x = [emb | ctx] directly in bf16 (only the gate GEMM consumes it).
__global__ void __launch_bounds__(128)
attn_kernel(const float* __restrict__ q,          // (B, DENC)
            const float* __restrict__ keys,       // (B*S, DENC)
            const float* __restrict__ enc,        // (B*S, DENC)
            const uint8_t* __restrict__ mask,     // (B, S) bool
            const float* __restrict__ emb_table,  // (V, E)
            const int* __restrict__ pred,
            unsigned short* __restrict__ xb)      // (B, E+DENC) bf16
{
    __shared__ float sq[DENC];
    __shared__ float sal[SS];
    __shared__ float red[128];
    const int b = blockIdx.x;
    const int t = threadIdx.x;

    for (int d = t; d < DENC; d += 128) sq[d] = q[b * DENC + d];
    __syncthreads();

    // one score per thread (S == 128)
    float sc = 0.f;
    {
        const float* kp = keys + ((size_t)b * SS + t) * DENC;
        for (int d = 0; d < DENC; ++d) sc += sq[d] * kp[d];
        sc *= 0.03125f;                       // 1/sqrt(1024)
        if (mask[b * SS + t]) sc = -1.0e10f;  // MASK_VALUE
    }
    red[t] = sc;
    __syncthreads();
    for (int o = 64; o > 0; o >>= 1) {
        if (t < o) red[t] = fmaxf(red[t], red[t + o]);
        __syncthreads();
    }
    const float mx = red[0];
    __syncthreads();
    const float ex = __expf(sc - mx);
    red[t] = ex;
    __syncthreads();
    for (int o = 64; o > 0; o >>= 1) {
        if (t < o) red[t] += red[t + o];
        __syncthreads();
    }
    sal[t] = ex / red[0];
    __syncthreads();

    // embedding half of x (bf16)
    const int p = pred[b];
    for (int e = t; e < EE; e += 128)
        xb[(size_t)b * (EE + DENC) + e] = f2bf_u(emb_table[(size_t)p * EE + e]);

    // context half of x (fp32 accumulate, bf16 store; coalesced over d)
    for (int d = t; d < DENC; d += 128) {
        float acc = 0.f;
        const float* ep = enc + (size_t)b * SS * DENC + d;
        for (int s = 0; s < SS; ++s) acc += sal[s] * ep[(size_t)s * DENC];
        xb[(size_t)b * (EE + DENC) + EE + d] = f2bf_u(acc);
    }
}

__global__ void lstm_act(const float* __restrict__ gates,  // (B, 4H) [i|f|g|o]
                         const float* __restrict__ b_hh,   // (4H)
                         float* __restrict__ h, float* __restrict__ c,
                         unsigned short* __restrict__ hb)  // bf16 copy for GEMM A
{
    int i = blockIdx.x * blockDim.x + threadIdx.x;
    if (i >= BB * HH) return;
    int b = i / HH, j = i % HH;
    const float* g = gates + (size_t)b * 4 * HH;
    float gi = g[j]           + b_hh[j];
    float gf = g[HH + j]      + b_hh[HH + j];
    float gg = g[2 * HH + j]  + b_hh[2 * HH + j];
    float go = g[3 * HH + j]  + b_hh[3 * HH + j];
    float i_ = 1.f / (1.f + __expf(-gi));
    float f_ = 1.f / (1.f + __expf(-gf));
    float g_ = tanhf(gg);
    float o_ = 1.f / (1.f + __expf(-go));
    float cn = f_ * c[i] + i_ * g_;
    float hn = o_ * tanhf(cn);
    c[i] = cn;
    h[i] = hn;
    hb[i] = f2bf_u(hn);
}

// per-row argmax (first-index tie-break like jnp.argmax) + scatter logits into
// the (B, V, TSTEPS) output slice for step t
__global__ void __launch_bounds__(256)
argmax_scatter(const float* __restrict__ logits,  // (B, V) compact
               float* __restrict__ out,           // (B, V, TSTEPS)
               int t, int* __restrict__ pred) {
    const int b = blockIdx.x, tid = threadIdx.x;
    const float* lp = logits + (size_t)b * VV;
    float best = -__builtin_inff();
    int bidx = 0;
    for (int v = tid; v < VV; v += 256) {
        float val = lp[v];
        out[((size_t)b * VV + v) * TSTEPS + t] = val;
        if (val > best) { best = val; bidx = v; }
    }
    __shared__ float sv[256];
    __shared__ int   si[256];
    sv[tid] = best; si[tid] = bidx;
    __syncthreads();
    for (int o = 128; o > 0; o >>= 1) {
        if (tid < o) {
            if (sv[tid + o] > sv[tid] ||
                (sv[tid + o] == sv[tid] && si[tid + o] < si[tid])) {
                sv[tid] = sv[tid + o]; si[tid] = si[tid + o];
            }
        }
        __syncthreads();
    }
    if (tid == 0) pred[b] = si[0];
}

static inline void cvt(const float* src, __bf16* dst, size_t n, hipStream_t s) {
    f32_to_bf16<<<dim3((unsigned)((n / 4 + 255) / 256)), 256, 0, s>>>(
        src, (unsigned short*)dst, (int)n);
}

extern "C" void kernel_launch(void* const* d_in, const int* in_sizes, int n_in,
                              void* d_out, int out_size, void* d_ws, size_t ws_size,
                              hipStream_t stream) {
    const float*   enc   = (const float*)d_in[0];
    const uint8_t* mask  = (const uint8_t*)d_in[1];
    const float*   emb   = (const float*)d_in[2];
    const float*   wq    = (const float*)d_in[3];
    const float*   bq    = (const float*)d_in[4];
    const float*   wk    = (const float*)d_in[5];
    const float*   bk    = (const float*)d_in[6];
    const float*   w_ih0 = (const float*)d_in[7];
    const float*   w_hh0 = (const float*)d_in[8];
    const float*   b_ih0 = (const float*)d_in[9];
    const float*   b_hh0 = (const float*)d_in[10];
    const float*   w_ih1 = (const float*)d_in[11];
    const float*   w_hh1 = (const float*)d_in[12];
    const float*   b_ih1 = (const float*)d_in[13];
    const float*   b_hh1 = (const float*)d_in[14];
    const float*   w_out = (const float*)d_in[15];
    const float*   b_out = (const float*)d_in[16];
    float* out = (float*)d_out;

    // ---- workspace layout: fp32 region, then bf16 region ----
    float* ws = (float*)d_ws;
    size_t off = 0;
    float* keys   = ws + off; off += (size_t)BB * SS * DENC;   // 8.4M f32
    float* q      = ws + off; off += (size_t)BB * DENC;
    float* gates  = ws + off; off += (size_t)BB * 4 * HH;
    float* h0     = ws + off; off += (size_t)BB * HH;
    float* c0     = ws + off; off += (size_t)BB * HH;
    float* h1     = ws + off; off += (size_t)BB * HH;
    float* c1     = ws + off; off += (size_t)BB * HH;
    float* logits = ws + off; off += (size_t)BB * VV;          // 2.0M f32
    int*   pred   = (int*)(ws + off); off += 64;

    __bf16* bfb = (__bf16*)(ws + off);
    size_t bo = 0;
    __bf16* encb  = bfb + bo; bo += (size_t)BB * SS * DENC;    // 8.4M bf16
    __bf16* wkb   = bfb + bo; bo += (size_t)DENC * DENC;
    __bf16* wqb   = bfb + bo; bo += (size_t)DENC * HH;
    __bf16* wih0b = bfb + bo; bo += (size_t)4 * HH * (EE + DENC);
    __bf16* whh0b = bfb + bo; bo += (size_t)4 * HH * HH;
    __bf16* wih1b = bfb + bo; bo += (size_t)4 * HH * HH;
    __bf16* whh1b = bfb + bo; bo += (size_t)4 * HH * HH;
    __bf16* woutb = bfb + bo; bo += (size_t)VV * HH;           // 16.4M bf16
    __bf16* xb    = bfb + bo; bo += (size_t)BB * (EE + DENC);
    __bf16* h0b   = bfb + bo; bo += (size_t)BB * HH;
    __bf16* h1b   = bfb + bo; bo += (size_t)BB * HH;

    // ---- one-time bf16 conversions (weights + encoder) ----
    cvt(enc,   encb,  (size_t)BB * SS * DENC, stream);
    cvt(wk,    wkb,   (size_t)DENC * DENC, stream);
    cvt(wq,    wqb,   (size_t)DENC * HH, stream);
    cvt(w_ih0, wih0b, (size_t)4 * HH * (EE + DENC), stream);
    cvt(w_hh0, whh0b, (size_t)4 * HH * HH, stream);
    cvt(w_ih1, wih1b, (size_t)4 * HH * HH, stream);
    cvt(w_hh1, whh1b, (size_t)4 * HH * HH, stream);
    cvt(w_out, woutb, (size_t)VV * HH, stream);

    init_state<<<dim3((BB * HH + 255) / 256), 256, 0, stream>>>(
        h0, c0, h1, c1, (unsigned short*)h0b, (unsigned short*)h1b, pred);

    // keys = enc @ wk^T + bk   (M=B*S=8192, N=1024, K=1024) — once
    gemm_bf16_wmma<<<dim3(DENC / 64, (BB * SS) / 64), 128, 0, stream>>>(
        encb, wkb, DENC, nullptr, nullptr, 0, bk, keys, DENC);

    for (int t = 0; t < TSTEPS; ++t) {
        // q = h1 @ wq^T + bq   (M=64, N=1024, K=512)
        gemm_bf16_wmma<<<dim3(DENC / 64, 1), 128, 0, stream>>>(
            h1b, wqb, HH, nullptr, nullptr, 0, bq, q, DENC);

        attn_kernel<<<dim3(BB), 128, 0, stream>>>(
            q, keys, enc, mask, emb, pred, (unsigned short*)xb);

        // layer-0 gates = x@w_ih0^T + h0@w_hh0^T + b_ih0   (N=2048, K=1536/512)
        gemm_bf16_wmma<<<dim3((4 * HH) / 64, 1), 128, 0, stream>>>(
            xb, wih0b, EE + DENC, h0b, whh0b, HH, b_ih0, gates, 4 * HH);
        lstm_act<<<dim3((BB * HH + 255) / 256), 256, 0, stream>>>(
            gates, b_hh0, h0, c0, (unsigned short*)h0b);

        // layer-1 gates = h0_new@w_ih1^T + h1_old@w_hh1^T + b_ih1
        gemm_bf16_wmma<<<dim3((4 * HH) / 64, 1), 128, 0, stream>>>(
            h0b, wih1b, HH, h1b, whh1b, HH, b_ih1, gates, 4 * HH);
        lstm_act<<<dim3((BB * HH + 255) / 256), 256, 0, stream>>>(
            gates, b_hh1, h1, c1, (unsigned short*)h1b);

        // logits = h1 @ w_out^T + b_out   (M=64, N=32000, K=512)
        gemm_bf16_wmma<<<dim3(VV / 64, 1), 128, 0, stream>>>(
            h1b, woutb, HH, nullptr, nullptr, 0, b_out, logits, VV);

        argmax_scatter<<<dim3(BB), 256, 0, stream>>>(logits, out, t, pred);
    }
}
